// attention_Module_75780402970803
// MI455X (gfx1250) — matrix operand order repaired and checked by reading, not compiled, not run on hardware
//
#include <hip/hip_runtime.h>
#include <hip/hip_bf16.h>
#include <math.h>

typedef __attribute__((ext_vector_type(16))) _Float16 v16h;
typedef __attribute__((ext_vector_type(8)))  float    v8f;
typedef __attribute__((ext_vector_type(2)))  _Float16 h2;

#define BDIM   4
#define CDIM   128
#define FDIM   64
#define TDIM   2048
#define CF     8192      // C*F
#define DQK    10        // ATT_DIM
#define QKOUT  4

// ---------------- workspace layout (bytes, 256-aligned regions) -------------
#define SZ_WEFF   (DQK * CF * sizeof(float))           // 327680
#define OFF_WQE   ((size_t)0)
#define OFF_WKE   (OFF_WQE + SZ_WEFF)
#define OFF_BQE   (OFF_WKE + SZ_WEFF)                  // 10 floats, pad 256
#define OFF_BKE   (OFF_BQE + 256)
#define OFF_WVH   (OFF_BKE + 256)                      // 128*128 f16 = 32768
#define OFF_QB    (OFF_WVH + (size_t)CDIM*CDIM*2)
#define SZ_QK     ((size_t)BDIM*TDIM*DQK*sizeof(float))
#define OFF_KB    (OFF_QB + SZ_QK)
#define OFF_ATT   (OFF_KB + SZ_QK)                     // B*T*T f16 = 32MB

// =============== kernel 1: effective weights / biases / f16 Wv ==============
__global__ void prep_kernel(const float* __restrict__ Wq_conv, const float* __restrict__ Wq_lin,
                            const float* __restrict__ bq_conv, const float* __restrict__ bq_lin,
                            const float* __restrict__ Wk_conv, const float* __restrict__ Wk_lin,
                            const float* __restrict__ bk_conv, const float* __restrict__ bk_lin,
                            const float* __restrict__ Wv_conv,
                            float* __restrict__ WeffQ, float* __restrict__ WeffK,
                            float* __restrict__ beffQ, float* __restrict__ beffK,
                            _Float16* __restrict__ Wvh) {
    int tid = blockIdx.x * blockDim.x + threadIdx.x;
    if (tid < DQK * CF) {
        int d = tid >> 13;          // /8192
        int n = tid & (CF - 1);
        int c = n >> 6;             // /64
        int f = n & 63;
        float sq = 0.f, sk = 0.f;
#pragma unroll
        for (int o = 0; o < QKOUT; ++o) {
            sq += Wq_lin[d * (QKOUT * FDIM) + o * FDIM + f] * Wq_conv[o * CDIM + c];
            sk += Wk_lin[d * (QKOUT * FDIM) + o * FDIM + f] * Wk_conv[o * CDIM + c];
        }
        WeffQ[tid] = sq;
        WeffK[tid] = sk;
    }
    if (tid < CDIM * CDIM) Wvh[tid] = (_Float16)Wv_conv[tid];
    if (tid < DQK) {
        float sq = bq_lin[tid], sk = bk_lin[tid];
        for (int o = 0; o < QKOUT; ++o)
            for (int f = 0; f < FDIM; ++f) {
                sq += bq_conv[o] * Wq_lin[tid * (QKOUT * FDIM) + o * FDIM + f];
                sk += bk_conv[o] * Wk_lin[tid * (QKOUT * FDIM) + o * FDIM + f];
            }
        beffQ[tid] = sq;
        beffK[tid] = sk;
    }
}

// =============== kernel 2: init q/k with effective bias =====================
__global__ void qk_init_kernel(const float* __restrict__ beffQ, const float* __restrict__ beffK,
                               float* __restrict__ qb, float* __restrict__ kb) {
    int idx = blockIdx.x * blockDim.x + threadIdx.x;
    if (idx < BDIM * TDIM * DQK) {
        int d = idx % DQK;
        qb[idx] = beffQ[d];
        kb[idx] = beffK[d];
    }
}

// =============== kernel 3: q/k projection (split-n partial sums) ============
__global__ void qk_partial_kernel(const float* __restrict__ x,
                                  const float* __restrict__ WeffQ, const float* __restrict__ WeffK,
                                  float* __restrict__ qb, float* __restrict__ kb) {
    int bt = blockIdx.x * 256 + threadIdx.x;   // 0..8191 -> (b,t)
    int b = bt >> 11;
    int t = bt & (TDIM - 1);
    int n0 = blockIdx.y * 256;                 // 32 splits of 256 over CF
    float aq[DQK], ak[DQK];
#pragma unroll
    for (int d = 0; d < DQK; ++d) { aq[d] = 0.f; ak[d] = 0.f; }
    const float* xb = x + (size_t)b * CF * TDIM + t;
    for (int nn = 0; nn < 256; ++nn) {
        int n = n0 + nn;
        float xv = xb[(size_t)n * TDIM];       // coalesced across lanes (t fast)
#pragma unroll
        for (int d = 0; d < DQK; ++d) {
            aq[d] = fmaf(xv, WeffQ[d * CF + n], aq[d]);   // uniform -> s_load
            ak[d] = fmaf(xv, WeffK[d * CF + n], ak[d]);
        }
    }
    float* qp = qb + (size_t)bt * DQK;
    float* kp = kb + (size_t)bt * DQK;
#pragma unroll
    for (int d = 0; d < DQK; ++d) {
        atomicAdd(qp + d, aq[d]);
        atomicAdd(kp + d, ak[d]);
    }
}

// =============== kernel 4: fused energy + softmax -> f16 attention ==========
// grid (T/8, B), block 256 = 8 waves, one row i per wave. All K rows in LDS.
__global__ void att_softmax_kernel(const float* __restrict__ qb, const float* __restrict__ kb,
                                   _Float16* __restrict__ att) {
    extern __shared__ float kl[];              // TDIM*DQK floats = 80KB
    int b = blockIdx.y;
    int wave = threadIdx.x >> 5;
    int lane = threadIdx.x & 31;
    int i = blockIdx.x * 8 + wave;

    for (int idx = threadIdx.x; idx < TDIM * DQK; idx += 256)
        kl[idx] = kb[(size_t)b * TDIM * DQK + idx];
    __syncthreads();

    float qr[DQK];
#pragma unroll
    for (int d = 0; d < DQK; ++d)
        qr[d] = qb[((size_t)b * TDIM + i) * DQK + d];

    float e[TDIM / 32];
    float m = -3.4e38f;
#pragma unroll
    for (int j0 = 0; j0 < TDIM / 32; ++j0) {
        int j = j0 * 32 + lane;
        const float* kp = kl + j * DQK;
        float s = 0.f;
#pragma unroll
        for (int d = 0; d < DQK; ++d) s = fmaf(qr[d], kp[d], s);
        e[j0] = s;
        m = fmaxf(m, s);
    }
#pragma unroll
    for (int off = 16; off >= 1; off >>= 1) m = fmaxf(m, __shfl_xor(m, off, 32));
    float sum = 0.f;
#pragma unroll
    for (int j0 = 0; j0 < TDIM / 32; ++j0) {
        e[j0] = __expf(e[j0] - m);
        sum += e[j0];
    }
#pragma unroll
    for (int off = 16; off >= 1; off >>= 1) sum += __shfl_xor(sum, off, 32);
    float inv = 1.0f / sum;

    _Float16* arow = att + ((size_t)b * TDIM + i) * TDIM;
#pragma unroll
    for (int j0 = 0; j0 < TDIM / 32; ++j0)
        arow[j0 * 32 + lane] = (_Float16)(e[j0] * inv);
}

// =============== WMMA fragment gathers (wave32 layouts, 16x16x32 f16) =======
// A-matrix 16x32: lanes 0-15 M=0..15 hold K {0..7,16..23}; lanes 16-31 K {8..15,24..31}
__device__ __forceinline__ v16h frag_A(const _Float16* base, int stride, int rowbase,
                                       int kbase, int lane) {
    int r = rowbase + (lane & 15);
    int khi = (lane & 16) ? 8 : 0;
    v16h a;
#pragma unroll
    for (int v = 0; v < 8; ++v) {
        int kk = kbase + ((v < 4) ? (khi + 2 * v) : (16 + khi + 2 * (v - 4)));
        h2 p = *(const h2*)(base + r * stride + kk);
        a[2 * v] = p.x;
        a[2 * v + 1] = p.y;
    }
    return a;
}
// B-matrix 32x16: N = lane&15; lanes 0-15 K {0..7,16..23}; lanes 16-31 K {8..15,24..31}.
// Memory holds B^T row-major: element(K,N) at base[N*stride + K].
__device__ __forceinline__ v16h frag_B(const _Float16* base, int stride, int colbase,
                                       int kbase, int lane) {
    int c = colbase + (lane & 15);
    int kg = (lane & 16) ? 8 : 0;
    v16h bv;
#pragma unroll
    for (int v = 0; v < 8; ++v) {
        int kk = kbase + ((v < 4) ? (kg + 2 * v) : (16 + kg + 2 * (v - 4)));
        h2 p = *(const h2*)(base + c * stride + kk);
        bv[2 * v] = p.x;
        bv[2 * v + 1] = p.y;
    }
    return bv;
}

// =============== kernel 5: fused GEMM1 (X·Aᵀ) + GEMM2 (Wv·Y) + bias =========
// grid (T/128 i-tiles, F, B), block 256 = 8 waves (4 M-groups x 2 N-groups).
// GEMM1: Y[c',i] = sum_j x[b,c',f,j] * att[b,i,j]   (M=128 c', N=128 i, K=2048 j)
// GEMM2: out[c,i] = sum_c' Wv[c,c'] * Y[c',i] + bv[c]   (row-sum of softmax == 1)
#define LDA 34     // padded f16 stride for 32-wide staging tiles
#define LDY 130    // padded f16 stride for 128-wide Y^T tile
__global__ void __launch_bounds__(256)
attn_apply_wmma_kernel(const float* __restrict__ x, const _Float16* __restrict__ att,
                       const _Float16* __restrict__ Wvh, const float* __restrict__ bv,
                       float* __restrict__ out) {
    __shared__ _Float16 xs[CDIM * LDA];   // x tile  [c'][j]  f16
    __shared__ _Float16 as_[CDIM * LDA];  // att tile [i][j]  f16 (= B^T rows)
    __shared__ _Float16 ys[CDIM * LDY];   // Y^T tile [i][c'] f16

    const int i0 = blockIdx.x * 128;
    const int f  = blockIdx.y;
    const int b  = blockIdx.z;
    const int lane = threadIdx.x & 31;
    const int wave = threadIdx.x >> 5;
    const int mb0 = (wave & 3) * 32;      // 2 M-subtiles per wave
    const int nb0 = (wave >> 2) * 64;     // 4 N-subtiles per wave

    const float*    xb = x   + (size_t)b * CF * TDIM + (size_t)f * TDIM;      // + c'*(F*T) + j
    const _Float16* ab = att + ((size_t)b * TDIM + i0) * TDIM;                // + iloc*T + j

    v8f acc[2][4] = {};

    const int row  = threadIdx.x >> 1;    // 0..127 staging row
    const int half = threadIdx.x & 1;     // 16-element half of the 32-wide tile

    for (int j0 = 0; j0 < TDIM; j0 += 32) {
        // ---- stage x (f32 -> f16) and attention (f16) tiles into LDS ----
        {
            const float* gp = xb + (size_t)row * (FDIM * TDIM) + j0 + half * 16;
            float vals[16];
            *(float4*)(vals + 0)  = ((const float4*)gp)[0];
            *(float4*)(vals + 4)  = ((const float4*)gp)[1];
            *(float4*)(vals + 8)  = ((const float4*)gp)[2];
            *(float4*)(vals + 12) = ((const float4*)gp)[3];
            _Float16* dx = xs + row * LDA + half * 16;
#pragma unroll
            for (int u = 0; u < 16; ++u) dx[u] = (_Float16)vals[u];

            const uint4* a4 = (const uint4*)(ab + (size_t)row * TDIM + j0 + half * 16);
            uint4 u0 = a4[0], u1 = a4[1];
            uint* da = (uint*)(as_ + row * LDA + half * 16);
            da[0] = u0.x; da[1] = u0.y; da[2] = u0.z; da[3] = u0.w;
            da[4] = u1.x; da[5] = u1.y; da[6] = u1.z; da[7] = u1.w;
        }
        __syncthreads();
        // ---- 2x4 WMMA tiles per wave ----
        v16h af[2], bf[4];
#pragma unroll
        for (int mi = 0; mi < 2; ++mi) af[mi] = frag_A(xs, LDA, mb0 + mi * 16, 0, lane);
#pragma unroll
        for (int ni = 0; ni < 4; ++ni) bf[ni] = frag_B(as_, LDA, nb0 + ni * 16, 0, lane);
#pragma unroll
        for (int mi = 0; mi < 2; ++mi)
#pragma unroll
            for (int ni = 0; ni < 4; ++ni)
                acc[mi][ni] = __builtin_amdgcn_wmma_f32_16x16x32_f16(
                    false, af[mi], false, bf[ni], (short)0, acc[mi][ni], false, false);
        __syncthreads();
    }

    // ---- spill Y^T (i-major) to LDS as f16 ----
    const int rofs = (lane & 16) ? 8 : 0;
#pragma unroll
    for (int mi = 0; mi < 2; ++mi)
#pragma unroll
        for (int ni = 0; ni < 4; ++ni) {
            int colY = nb0 + ni * 16 + (lane & 15);
            int mrow = mb0 + mi * 16 + rofs;
#pragma unroll
            for (int r = 0; r < 8; ++r)
                ys[colY * LDY + mrow + r] = (_Float16)acc[mi][ni][r];
        }
    __syncthreads();

    // ---- GEMM2: out_tile = Wv (f16) x Y, K = 128 channels ----
    v8f acc2[2][4] = {};
#pragma unroll
    for (int kb = 0; kb < CDIM; kb += 32) {
        v16h aw[2], by[4];
#pragma unroll
        for (int mi = 0; mi < 2; ++mi) aw[mi] = frag_A(Wvh, CDIM, mb0 + mi * 16, kb, lane);
#pragma unroll
        for (int ni = 0; ni < 4; ++ni) by[ni] = frag_B(ys, LDY, nb0 + ni * 16, kb, lane);
#pragma unroll
        for (int mi = 0; mi < 2; ++mi)
#pragma unroll
            for (int ni = 0; ni < 4; ++ni)
                acc2[mi][ni] = __builtin_amdgcn_wmma_f32_16x16x32_f16(
                    false, aw[mi], false, by[ni], (short)0, acc2[mi][ni], false, false);
    }

    // ---- epilogue: + bv[c] (softmax rows sum to 1), store out[b,c,f,i] ----
    float* ob = out + (size_t)b * CDIM * FDIM * TDIM + (size_t)f * TDIM;
#pragma unroll
    for (int mi = 0; mi < 2; ++mi)
#pragma unroll
        for (int ni = 0; ni < 4; ++ni) {
            int icol = i0 + nb0 + ni * 16 + (lane & 15);
#pragma unroll
            for (int r = 0; r < 8; ++r) {
                int c = mb0 + mi * 16 + rofs + r;
                ob[(size_t)c * (FDIM * TDIM) + icol] = acc2[mi][ni][r] + bv[c];
            }
        }
}

// ============================== launcher ====================================
extern "C" void kernel_launch(void* const* d_in, const int* in_sizes, int n_in,
                              void* d_out, int out_size, void* d_ws, size_t ws_size,
                              hipStream_t stream) {
    const float* x       = (const float*)d_in[0];
    const float* Wq_conv = (const float*)d_in[1];
    const float* bq_conv = (const float*)d_in[2];
    const float* Wq_lin  = (const float*)d_in[3];
    const float* bq_lin  = (const float*)d_in[4];
    const float* Wk_conv = (const float*)d_in[5];
    const float* bk_conv = (const float*)d_in[6];
    const float* Wk_lin  = (const float*)d_in[7];
    const float* bk_lin  = (const float*)d_in[8];
    const float* Wv_conv = (const float*)d_in[9];
    const float* bv_conv = (const float*)d_in[10];
    float* out = (float*)d_out;

    char* w = (char*)d_ws;
    float*    WeffQ = (float*)(w + OFF_WQE);
    float*    WeffK = (float*)(w + OFF_WKE);
    float*    beffQ = (float*)(w + OFF_BQE);
    float*    beffK = (float*)(w + OFF_BKE);
    _Float16* Wvh   = (_Float16*)(w + OFF_WVH);
    float*    qb    = (float*)(w + OFF_QB);
    float*    kb    = (float*)(w + OFF_KB);
    _Float16* att   = (_Float16*)(w + OFF_ATT);

    prep_kernel<<<(DQK * CF + 255) / 256, 256, 0, stream>>>(
        Wq_conv, Wq_lin, bq_conv, bq_lin, Wk_conv, Wk_lin, bk_conv, bk_lin,
        Wv_conv, WeffQ, WeffK, beffQ, beffK, Wvh);

    qk_init_kernel<<<(BDIM * TDIM * DQK + 255) / 256, 256, 0, stream>>>(beffQ, beffK, qb, kb);

    qk_partial_kernel<<<dim3(BDIM * TDIM / 256, CF / 256), 256, 0, stream>>>(
        x, WeffQ, WeffK, qb, kb);

    att_softmax_kernel<<<dim3(TDIM / 8, BDIM), 256, TDIM * DQK * sizeof(float), stream>>>(
        qb, kb, att);

    attn_apply_wmma_kernel<<<dim3(TDIM / 128, FDIM, BDIM), 256, 0, stream>>>(
        x, att, Wvh, bv_conv, out);
}
